// VectorQuantizedVAE_35983236006238
// MI455X (gfx1250) — compile-verified, matrix operand order checked by reading, not run on hardware
//
#include <hip/hip_runtime.h>

// ---------------------------------------------------------------------------
// VQ-VAE forward for MI455X (gfx1250, wave32). All convs run as implicit
// GEMMs on v_wmma_f32_16x16x32_f16 with 2x4 register blocking (8 WMMAs per
// K-chunk per wave); VQ nearest-neighbor search also uses WMMA.
// ---------------------------------------------------------------------------

typedef __attribute__((ext_vector_type(16))) _Float16 v16h;
typedef __attribute__((ext_vector_type(8)))  float    v8f;

// K index inside a 16x32 f16 A/B fragment for half-element h (0..15) and lane.
// Per CDNA5 ISA 7.12.2: lanes 0-15 hold K {0..7,16..23}, lanes 16-31 hold
// K {8..15,24..31}; element h -> VGPR h/2, slot h%2.
__device__ __forceinline__ int kmap16(int h, int lane) {
  return (h & 7) + ((h >> 3) << 4) + (((lane >> 4) & 1) << 3);
}

// ---------------------------------------------------------------------------
// Implicit-GEMM conv1d.  X: [Cin][B=16][Lin]   W: [Cout][Cin][KW]
// Out: [Cout][16][Lout] (outBCL=0)  or  [16][Cout][Lout] (outBCL=1)
// One wave -> 32(Cout) x 64(position) macro-tile = 2x4 WMMA tiles.
// K loop = Cin*KW in steps of 32 (all layers: Cin*KW % 32 == 0).
// ---------------------------------------------------------------------------
template<int KW>
__global__ void conv_wmma(const float* __restrict__ X, const float* __restrict__ W,
                          const float* __restrict__ Bias, float* __restrict__ Out,
                          int Cin, int Cout, int Lin, int Lout,
                          int stride, int pad, int outBCL)
{
  const int lane = threadIdx.x & 31;
  const int wave = threadIdx.x >> 5;
  const int co0  = blockIdx.x * 32;                // 2 Cout tiles of 16
  const int n0   = (blockIdx.y * 4 + wave) * 64;   // 4 position tiles of 16
  const int b    = n0 / Lout;                      // Lout % 64 == 0 everywhere
  const int l0   = n0 - b * Lout;
  const int m    = lane & 15;                      // A-row (cout) / B-col (pos)
  const int Ktot = Cin * KW;

  v8f acc[2][4] = {};
  for (int kb = 0; kb < Ktot; kb += 32) {
    v16h afrag[2], bfrag[4];
#pragma unroll
    for (int h = 0; h < 16; ++h) {
      const int k  = kb + kmap16(h, lane);
      const int ci = k / KW;
      const int kw = k - ci * KW;
#pragma unroll
      for (int mt = 0; mt < 2; ++mt)               // weight fragments (A)
        afrag[mt][h] = (_Float16)W[((co0 + mt * 16 + m) * Cin + ci) * KW + kw];
      const float* Xrow = X + ((size_t)ci * 16 + b) * Lin;
#pragma unroll
      for (int nt = 0; nt < 4; ++nt) {             // activation fragments (B)
        const int pos = (l0 + nt * 16 + m) * stride + kw - pad;
        float xv = 0.0f;
        if (pos >= 0 && pos < Lin) xv = Xrow[pos];
        bfrag[nt][h] = (_Float16)xv;
      }
    }
#pragma unroll
    for (int mt = 0; mt < 2; ++mt)
#pragma unroll
      for (int nt = 0; nt < 4; ++nt)
        acc[mt][nt] = __builtin_amdgcn_wmma_f32_16x16x32_f16(
            false, afrag[mt], false, bfrag[nt], (short)0, acc[mt][nt], false, false);
  }
#pragma unroll
  for (int mt = 0; mt < 2; ++mt) {
#pragma unroll
    for (int nt = 0; nt < 4; ++nt) {
#pragma unroll
      for (int r = 0; r < 8; ++r) {
        const int mo = r + ((lane >> 4) << 3);     // D layout: vgpr r + lane-half
        const int n  = (lane & 15) + nt * 16;
        const int co = co0 + mt * 16 + mo;
        const float v = acc[mt][nt][r] + Bias[co];
        if (outBCL) Out[((size_t)b * Cout + co) * Lout + l0 + n] = v;
        else        Out[((size_t)co * 16 + b) * Lout + l0 + n]   = v;
      }
    }
  }
}

// ---------------------------------------------------------------------------
// Small helpers
// ---------------------------------------------------------------------------
__global__ void transpose_bcl_cbl(const float* __restrict__ x, float* __restrict__ xt)
{ // x: [16][64][4096] -> xt: [64][16][4096]
  const int i = blockIdx.x * 256 + threadIdx.x;
  const int b = i >> 18, c = (i >> 12) & 63, l = i & 4095;
  xt[((c << 4) + b) * 4096 + l] = x[i];
}

__global__ void zero_hist(float* __restrict__ hist)
{ const int i = blockIdx.x * 256 + threadIdx.x; if (i < 1024) hist[i] = 0.0f; }

__global__ void codebook_norms(const float* __restrict__ cb, float* __restrict__ cn)
{
  const int k = blockIdx.x * 256 + threadIdx.x;
  if (k < 1024) {
    float s = 0.0f;
    for (int d = 0; d < 128; ++d) { const float v = cb[k * 128 + d]; s += v * v; }
    cn[k] = s;
  }
}

__global__ void bn_stats(const float* __restrict__ xb, float* __restrict__ mean,
                         float* __restrict__ invstd, int N)
{
  const int c = blockIdx.x;
  const float* p = xb + (size_t)c * N;
  float s = 0.0f, ss = 0.0f;
  for (int i = threadIdx.x; i < N; i += 256) { const float v = p[i]; s += v; ss += v * v; }
  __shared__ float s1[256], s2[256];
  s1[threadIdx.x] = s; s2[threadIdx.x] = ss; __syncthreads();
  for (int o = 128; o > 0; o >>= 1) {
    if (threadIdx.x < o) { s1[threadIdx.x] += s1[threadIdx.x + o]; s2[threadIdx.x] += s2[threadIdx.x + o]; }
    __syncthreads();
  }
  if (threadIdx.x == 0) {
    const float m = s1[0] / (float)N;
    const float var = s2[0] / (float)N - m * m;       // biased variance (torch)
    mean[c] = m; invstd[c] = rsqrtf(var + 1e-5f);
  }
}

__global__ void bn_apply_relu(float* __restrict__ xb, const float* __restrict__ mean,
                              const float* __restrict__ invstd, const float* __restrict__ g,
                              const float* __restrict__ be, int N)
{
  const int i = blockIdx.x * 256 + threadIdx.x;
  const int c = i / N;
  const float v = g[c] * (xb[i] - mean[c]) * invstd[c] + be[c];
  xb[i] = fmaxf(v, 0.0f);
}

__global__ void upsample2(const float* __restrict__ in, float* __restrict__ out, int L)
{ // per (c,b) row: length L -> 2L nearest-neighbor
  const int i = blockIdx.x * 256 + threadIdx.x;
  const int row = i / (2 * L);
  const int t   = i - row * 2 * L;
  out[i] = in[row * L + (t >> 1)];
}

// ---------------------------------------------------------------------------
// VQ nearest-neighbor via WMMA.  Z: [128][4096] (d-major), CB: [1024][128].
// dist(n,k) = |c_k|^2 - 2 * <z_n, c_k>   (|z_n|^2 constant in argmin)
// Block = 4 waves; each wave owns 16 of the 64 code tiles for one 16-pos tile;
// the z fragments stay in registers across all code tiles.
// ---------------------------------------------------------------------------
__global__ void vq_argmin(const float* __restrict__ Z, const float* __restrict__ CB,
                          const float* __restrict__ cn, int* __restrict__ idx,
                          float* __restrict__ hist)
{
  const int lane = threadIdx.x & 31;
  const int wave = threadIdx.x >> 5;
  const int n0   = blockIdx.x * 16;
  const int m    = lane & 15;

  v16h zb[4];                                  // B fragments for K = 0..127
#pragma unroll
  for (int kc = 0; kc < 4; ++kc)
#pragma unroll
    for (int h = 0; h < 16; ++h) {
      const int d = kc * 32 + kmap16(h, lane);
      zb[kc][h] = (_Float16)Z[d * 4096 + n0 + m];
    }

  float best = 3.0e38f; int bidx = 0;
  for (int kt = wave; kt < 64; kt += 4) {      // 16 code tiles per wave
    v8f g = {};
#pragma unroll
    for (int kc = 0; kc < 4; ++kc) {
      v16h a;
#pragma unroll
      for (int h = 0; h < 16; ++h) {
        const int d = kc * 32 + kmap16(h, lane);
        a[h] = (_Float16)CB[(kt * 16 + m) * 128 + d];
      }
      g = __builtin_amdgcn_wmma_f32_16x16x32_f16(
          false, a, false, zb[kc], (short)0, g, false, false);
    }
#pragma unroll
    for (int r = 0; r < 8; ++r) {
      const int code = kt * 16 + r + ((lane >> 4) << 3);
      const float dist = cn[code] - 2.0f * g[r];
      if (dist < best || (dist == best && code < bidx)) { best = dist; bidx = code; }
    }
  }
  __shared__ float sval[4][32];
  __shared__ int   sidx[4][32];
  sval[wave][lane] = best; sidx[wave][lane] = bidx;
  __syncthreads();
  if (threadIdx.x < 16) {                      // each thread: one position n
    const int n = threadIdx.x;
    float bv = 3.0e38f; int bi = 0x7fffffff;
    for (int w2 = 0; w2 < 4; ++w2)
      for (int hh = 0; hh < 2; ++hh) {
        const float v = sval[w2][n + 16 * hh];
        const int  ii = sidx[w2][n + 16 * hh];
        if (v < bv || (v == bv && ii < bi)) { bv = v; bi = ii; }
      }
    idx[n0 + n] = bi;
    atomicAdd(&hist[bi], 1.0f);                // exact small-int fp32 adds
  }
}

// z_q gather + z_e/z_q relayout + per-block partial of sum((z_q-z_e)^2)
__global__ void vq_gather_loss(const float* __restrict__ ze, const float* __restrict__ cb,
                               const int* __restrict__ idx, float* __restrict__ out_ze,
                               float* __restrict__ out_zq, float* __restrict__ zq_cbl,
                               float* __restrict__ partial)
{
  const int i = blockIdx.x * 256 + threadIdx.x;   // i = c*4096 + n, 524288 total
  const int c = i >> 12, n = i & 4095;
  const int b = n >> 8,  l = n & 255;
  const float zev = ze[i];
  const float zqv = cb[idx[n] * 128 + c];
  out_ze[((size_t)b * 128 + c) * 256 + l] = zev;  // [B][C][L]
  out_zq[((size_t)b * 128 + c) * 256 + l] = zqv;
  zq_cbl[i] = zqv;                                // [C][B][L] for decoder
  float d = zqv - zev; d *= d;
  __shared__ float s[256];
  s[threadIdx.x] = d; __syncthreads();
  for (int o = 128; o > 0; o >>= 1) {
    if (threadIdx.x < o) s[threadIdx.x] += s[threadIdx.x + o];
    __syncthreads();
  }
  if (threadIdx.x == 0) partial[blockIdx.x] = s[0];
}

__global__ void finalize_loss(const float* __restrict__ partial, float* __restrict__ loss)
{
  float a = 0.0f;
  for (int i = threadIdx.x; i < 2048; i += 256) a += partial[i];
  __shared__ float s[256];
  s[threadIdx.x] = a; __syncthreads();
  for (int o = 128; o > 0; o >>= 1) {
    if (threadIdx.x < o) s[threadIdx.x] += s[threadIdx.x + o];
    __syncthreads();
  }
  // vq_loss = codebook_loss + BETA*commitment_loss = 1.25 * mean(sq)
  if (threadIdx.x == 0) loss[0] = s[0] * (1.25f / 524288.0f);
}

__global__ void perplexity_k(const float* __restrict__ hist, float* __restrict__ out)
{
  float a = 0.0f;
  for (int i = threadIdx.x; i < 1024; i += 256) {
    const float p = hist[i] * (1.0f / 4096.0f);
    a += p * logf(p + 1e-10f);
  }
  __shared__ float s[256];
  s[threadIdx.x] = a; __syncthreads();
  for (int o = 128; o > 0; o >>= 1) {
    if (threadIdx.x < o) s[threadIdx.x] += s[threadIdx.x + o];
    __syncthreads();
  }
  if (threadIdx.x == 0) out[0] = expf(-s[0]);
}

__global__ void write_idx(const int* __restrict__ idx, int* __restrict__ out)
{ const int i = blockIdx.x * 256 + threadIdx.x; out[i] = idx[i]; }

// ---------------------------------------------------------------------------
// Host-side orchestration (graph-capture safe: launches only)
// ---------------------------------------------------------------------------
extern "C" void kernel_launch(void* const* d_in, const int* in_sizes, int n_in,
                              void* d_out, int out_size, void* d_ws, size_t ws_size,
                              hipStream_t stream)
{
  (void)in_sizes; (void)n_in; (void)out_size; (void)ws_size;
  const float* x  = (const float*)d_in[0];
  const float* cb = (const float*)d_in[45];
  float* outF = (float*)d_out;

  // Workspace layout (floats). Needs ~286 MB.
  float* A      = (float*)d_ws;                       // 512*16*4096
  float* Bb     = A  + (size_t)512 * 16 * 4096;       // 512*16*4096
  float* XT     = Bb + (size_t)512 * 16 * 4096;       // 64*16*4096
  float* mean   = XT + (size_t)64 * 16 * 4096;        // 512
  float* invstd = mean + 512;                         // 512
  float* cnorm  = invstd + 512;                       // 1024
  int*   idxb   = (int*)(cnorm + 1024);               // 4096
  float* hist   = (float*)(idxb + 4096);              // 1024
  float* part   = hist + 1024;                        // 2048

  const size_t off_ze = 4194304, off_zq = off_ze + 524288,
               off_loss = off_zq + 524288, off_perp = off_loss + 1,
               off_idx = off_perp + 1;

  transpose_bcl_cbl<<<(16 * 64 * 4096) / 256, 256, 0, stream>>>(x, XT);
  zero_hist<<<4, 256, 0, stream>>>(hist);
  codebook_norms<<<4, 256, 0, stream>>>(cb, cnorm);

  // ---- encoder ----
  const int encC[5] = {64, 512, 256, 128, 128};
  int L = 4096;
  const float* cur = XT;
  for (int i = 0; i < 4; ++i) {
    const int Ci = encC[i], Co = encC[i + 1];
    const float* w1 = (const float*)d_in[1 + 6 * i + 0];
    const float* b1 = (const float*)d_in[1 + 6 * i + 1];
    const float* g  = (const float*)d_in[1 + 6 * i + 2];
    const float* be = (const float*)d_in[1 + 6 * i + 3];
    const float* w2 = (const float*)d_in[1 + 6 * i + 4];
    const float* b2 = (const float*)d_in[1 + 6 * i + 5];
    conv_wmma<3><<<dim3(Co / 32, L / 16), 128, 0, stream>>>(
        cur, w1, b1, A, Ci, Co, L, L, 1, 1, 0);
    bn_stats<<<Co, 256, 0, stream>>>(A, mean, invstd, 16 * L);
    bn_apply_relu<<<(Co * 16 * L) / 256, 256, 0, stream>>>(A, mean, invstd, g, be, 16 * L);
    conv_wmma<3><<<dim3(Co / 32, (L / 2) / 16), 128, 0, stream>>>(
        A, w2, b2, Bb, Co, Co, L, L / 2, 2, 1, 0);
    cur = Bb; L /= 2;
  }
  // 1x1 projection -> z_e in A  (L == 256)
  conv_wmma<1><<<dim3(128 / 32, L / 16), 128, 0, stream>>>(
      Bb, (const float*)d_in[25], (const float*)d_in[26], A, 128, 128, L, L, 1, 0, 0);

  // ---- quantize ----
  vq_argmin<<<256, 128, 0, stream>>>(A, cb, cnorm, idxb, hist);
  vq_gather_loss<<<2048, 256, 0, stream>>>(A, cb, idxb, outF + off_ze, outF + off_zq, Bb, part);
  finalize_loss<<<1, 256, 0, stream>>>(part, outF + off_loss);
  perplexity_k<<<1, 256, 0, stream>>>(hist, outF + off_perp);
  write_idx<<<16, 256, 0, stream>>>(idxb, (int*)d_out + off_idx);

  // ---- decoder ---- (z_q already in Bb as [128][16][256])
  const int decC[5] = {128, 64, 128, 256, 512};
  L = 256;
  for (int i = 0; i < 4; ++i) {
    const int Ci = decC[i], Co = decC[i + 1];
    const float* w  = (const float*)d_in[27 + 4 * i + 0];
    const float* b  = (const float*)d_in[27 + 4 * i + 1];
    const float* g  = (const float*)d_in[27 + 4 * i + 2];
    const float* be = (const float*)d_in[27 + 4 * i + 3];
    conv_wmma<3><<<dim3(Co / 32, L / 16), 128, 0, stream>>>(
        Bb, w, b, A, Ci, Co, L, L, 1, 1, 0);
    bn_stats<<<Co, 256, 0, stream>>>(A, mean, invstd, 16 * L);
    bn_apply_relu<<<(Co * 16 * L) / 256, 256, 0, stream>>>(A, mean, invstd, g, be, 16 * L);
    upsample2<<<(Co * 16 * 2 * L) / 256, 256, 0, stream>>>(A, Bb, L);
    L *= 2;
  }
  // final conv -> x_hat [16][64][4096] directly into d_out
  conv_wmma<3><<<dim3(64 / 32, 4096 / 16), 128, 0, stream>>>(
      Bb, (const float*)d_in[43], (const float*)d_in[44], outF, 512, 64, 4096, 4096, 1, 1, 1);
}
